// RevRNN_76897094468454
// MI455X (gfx1250) — compile-verified, matrix operand order checked by reading, not run on hardware
//
#include <hip/hip_runtime.h>
#include <hip/hip_bf16.h>
#include <cmath>

typedef _Float16 v16h __attribute__((ext_vector_type(16)));
typedef _Float16 v8h  __attribute__((ext_vector_type(8)));
typedef float    v8f  __attribute__((ext_vector_type(8)));

#define B_SZ   256      // batch
#define HID    512      // h
#define TWOH   1024     // 2h
#define IN_F   256      // input features
#define KDIM   770      // I + 2 + h
#define KPAD   800      // K padded to multiple of 32
#define SEQ    512      // base sequence length
#define STEPS  520      // SEQ + DELAY
#define EPS    1e-6f

// ---------------------------------------------------------------------------
// carry[b, j] = hidden_state[j]   (carry lives in d_out, [256, 1024] f32)
// ---------------------------------------------------------------------------
__global__ void k_init_carry(float* __restrict__ carry,
                             const float* __restrict__ hidden) {
    int i = blockIdx.x * blockDim.x + threadIdx.x;    // 256*1024 threads
    carry[i] = hidden[i & (TWOH - 1)];
}

// ---------------------------------------------------------------------------
// f32 [1024, 770] weights -> f16 [1024, 800] (zero-padded K)
// ---------------------------------------------------------------------------
__global__ void k_cvt_w(const float* __restrict__ w, _Float16* __restrict__ o) {
    int i = blockIdx.x * blockDim.x + threadIdx.x;    // 1024*800 threads
    int r = i / KPAD, c = i - r * KPAD;
    o[i] = (c < KDIM) ? (_Float16)w[r * KDIM + c] : (_Float16)0.f;
}

// ---------------------------------------------------------------------------
// Build f16 A-matrix [256, 800] for one _calc:
//   cols   0..511 : mish( batchnorm( carry[:, half_off + f] ) )
//   cols 512..767 : fn_input[:, t, :]  (zero for t >= 512)
//   col  768      : idx = t+1
//   col  769      : (idx - factor) / factor,  factor = (S+1)/2
//   cols 770..799 : 0
// grid = 800 blocks (one per column), 256 threads (one per batch row)
// ---------------------------------------------------------------------------
__global__ void k_pack(const float* __restrict__ carry, int half_off,
                       const float* __restrict__ fn_in,
                       const float* __restrict__ bw, const float* __restrict__ bb,
                       _Float16* __restrict__ A, int t) {
    __shared__ float s1[B_SZ], s2[B_SZ];
    int f = blockIdx.x;
    int b = threadIdx.x;
    if (f < HID) {
        float x = carry[b * TWOH + half_off + f];
        s1[b] = x;
        s2[b] = x * x;
        __syncthreads();
        #pragma unroll
        for (int s = B_SZ / 2; s > 0; s >>= 1) {
            if (b < s) { s1[b] += s1[b + s]; s2[b] += s2[b + s]; }
            __syncthreads();
        }
        float mean = s1[0] * (1.f / B_SZ);
        float var  = (s2[0] - (float)B_SZ * mean * mean) * (1.f / (B_SZ - 1));
        var = var < 0.f ? 0.f : var;
        float o  = (x - mean) / (sqrtf(var) + EPS) * bw[f] + bb[f];
        float sp = (o > 20.f) ? o : log1pf(__expf(o));     // softplus
        float m  = o * tanhf(sp);                          // mish
        A[b * KPAD + f] = (_Float16)m;
    } else {
        int c = f;                                         // 512..799
        float v;
        if (c < HID + IN_F) {
            v = (t < SEQ) ? fn_in[((size_t)b * SEQ + t) * IN_F + (c - HID)] : 0.f;
        } else if (c == HID + IN_F) {
            v = (float)(t + 1);
        } else if (c == HID + IN_F + 1) {
            const float factor = (float)(STEPS + 1) * 0.5f;
            v = ((float)(t + 1) - factor) / factor;
        } else {
            v = 0.f;
        }
        A[b * KPAD + c] = (_Float16)v;
    }
}

// ---------------------------------------------------------------------------
// 16-bit WMMA fragment loader (A and B use the same per-lane pattern):
// lane L holds row (L&15); lanes 0-15 cover K {0..7, 16..23},
// lanes 16-31 cover K {8..15, 24..31}  -> two contiguous b128 loads.
// ---------------------------------------------------------------------------
__device__ __forceinline__ v16h load_frag(const _Float16* __restrict__ base,
                                          int row0, int kt, int lane) {
    const _Float16* p = base + (size_t)(row0 + (lane & 15)) * KPAD
                             + kt + ((lane & 16) ? 8 : 0);
    v8h lo = *(const v8h*)p;         // K = koff .. koff+7
    v8h hi = *(const v8h*)(p + 16);  // K = koff+16 .. koff+23
    return __builtin_shufflevector(lo, hi,
        0, 1, 2, 3, 4, 5, 6, 7, 8, 9, 10, 11, 12, 13, 14, 15);
}

#define WMMA_F16(a, b, c) __builtin_amdgcn_wmma_f32_16x16x32_f16( \
        false, (a), false, (b), (short)0, (c), false, false)

// ---------------------------------------------------------------------------
// z[256,1024] = A[256,800] @ W[1024,800]^T, then fused epilogue:
//   Cio[m, n] = z[m, n] * tanh(z[m, n+512]) + Cio[m, n]     (n in 0..511)
// Block: 128 thr = 4 waves (2 row x 2 col); block tile 64x64 per z-half.
// Wave tile: 32 rows x 32 cols of each half -> 8 accumulators, 8 WMMA/K-step
// from 6 fragments (2 A + 4 B). Software-pipelined: K-step k+1 fragment
// loads are issued before the WMMAs of K-step k, so WMMAs wait on loads
// that have a full iteration of latency already absorbed.
// ---------------------------------------------------------------------------
__global__ void __launch_bounds__(128)
k_gemm(const _Float16* __restrict__ A, const _Float16* __restrict__ W,
       float* __restrict__ Cio) {
    int lane = threadIdx.x & 31;
    int wv   = threadIdx.x >> 5;     // 0..3
    int wr   = wv >> 1;              // 0..1
    int wc   = wv & 1;               // 0..1
    int m0   = blockIdx.x * 64 + wr * 32;
    int n0   = blockIdx.y * 64 + wc * 32;

    v8f acc0[2][2] = {};             // [row subtile][col subtile], o0 half
    v8f acc1[2][2] = {};             // o1 half (W rows 512+n)

    // prologue: fragments for kt = 0
    v16h a[2], b0[2], b1[2];
    #pragma unroll
    for (int i = 0; i < 2; ++i) {
        a[i]  = load_frag(A, m0 + i * 16, 0, lane);
        b0[i] = load_frag(W, n0 + i * 16, 0, lane);
        b1[i] = load_frag(W, HID + n0 + i * 16, 0, lane);
    }

    #pragma unroll 2
    for (int kt = 32; kt < KPAD; kt += 32) {
        // issue next K-step loads first (pipeline depth 1)
        v16h an[2], b0n[2], b1n[2];
        #pragma unroll
        for (int i = 0; i < 2; ++i) {
            an[i]  = load_frag(A, m0 + i * 16, kt, lane);
            b0n[i] = load_frag(W, n0 + i * 16, kt, lane);
            b1n[i] = load_frag(W, HID + n0 + i * 16, kt, lane);
        }
        // 8 WMMAs on the previous K-step's fragments
        #pragma unroll
        for (int rr = 0; rr < 2; ++rr) {
            #pragma unroll
            for (int cc = 0; cc < 2; ++cc) {
                acc0[rr][cc] = WMMA_F16(a[rr], b0[cc], acc0[rr][cc]);
                acc1[rr][cc] = WMMA_F16(a[rr], b1[cc], acc1[rr][cc]);
            }
        }
        #pragma unroll
        for (int i = 0; i < 2; ++i) { a[i] = an[i]; b0[i] = b0n[i]; b1[i] = b1n[i]; }
    }
    // drain: last K-step
    #pragma unroll
    for (int rr = 0; rr < 2; ++rr) {
        #pragma unroll
        for (int cc = 0; cc < 2; ++cc) {
            acc0[rr][cc] = WMMA_F16(a[rr], b0[cc], acc0[rr][cc]);
            acc1[rr][cc] = WMMA_F16(a[rr], b1[cc], acc1[rr][cc]);
        }
    }

    // C/D layout: lane (0-15 | 16-31) -> col (lane&15); VGPR j -> row j (+8)
    #pragma unroll
    for (int rr = 0; rr < 2; ++rr) {
        #pragma unroll
        for (int cc = 0; cc < 2; ++cc) {
            int col   = n0 + cc * 16 + (lane & 15);
            int rbase = m0 + rr * 16 + ((lane & 16) ? 8 : 0);
            #pragma unroll
            for (int j = 0; j < 8; ++j) {
                float z0 = acc0[rr][cc][j];
                float z1 = acc1[rr][cc][j];
                float* p = Cio + (size_t)(rbase + j) * TWOH + col;
                *p = z0 * tanhf(z1) + *p;
            }
        }
    }
}

// ---------------------------------------------------------------------------
extern "C" void kernel_launch(void* const* d_in, const int* in_sizes, int n_in,
                              void* d_out, int out_size, void* d_ws, size_t ws_size,
                              hipStream_t stream) {
    const float* fn_input = (const float*)d_in[0];   // [256, 512, 256]
    const float* lp0      = (const float*)d_in[1];   // [1024, 770]
    const float* lp1      = (const float*)d_in[2];   // [1024, 770]
    const float* bw0      = (const float*)d_in[3];   // [512]
    const float* bb0      = (const float*)d_in[4];   // [512]
    const float* bw1      = (const float*)d_in[5];   // [512]
    const float* hidden   = (const float*)d_in[6];   // [1024]

    float* carry = (float*)d_out;                    // [256, 1024] == output

    char* ws = (char*)d_ws;
    _Float16* W0h  = (_Float16*)(ws);
    _Float16* W1h  = (_Float16*)(ws + (size_t)TWOH * KPAD * sizeof(_Float16));
    _Float16* Abuf = (_Float16*)(ws + (size_t)TWOH * KPAD * sizeof(_Float16) * 2);

    k_init_carry<<<(B_SZ * TWOH) / 256, 256, 0, stream>>>(carry, hidden);
    k_cvt_w<<<(TWOH * KPAD) / 256, 256, 0, stream>>>(lp0, W0h);
    k_cvt_w<<<(TWOH * KPAD) / 256, 256, 0, stream>>>(lp1, W1h);

    dim3 ggrid(B_SZ / 64, HID / 64);                 // 4 x 8 blocks
    for (int t = 0; t < STEPS; ++t) {
        // o1 = calc(inp0, s_t, w0, b0, P0) + inp1   -> carry[:, 512:]
        k_pack<<<KPAD, B_SZ, 0, stream>>>(carry, 0, fn_input, bw0, bb0, Abuf, t);
        k_gemm<<<ggrid, 128, 0, stream>>>(Abuf, W0h, carry + HID);
        // o0 = calc(o1, s_t, w1, w1, P1) + inp0     -> carry[:, :512]
        k_pack<<<KPAD, B_SZ, 0, stream>>>(carry, HID, fn_input, bw1, bw1, Abuf, t);
        k_gemm<<<ggrid, 128, 0, stream>>>(Abuf, W1h, carry);
    }
}
